// PhotonicDelayReservoir_46162308497411
// MI455X (gfx1250) — compile-verified, matrix operand order checked by reading, not run on hardware
//
#include <hip/hip_runtime.h>
#include <hip/hip_bf16.h>

// ---------------------------------------------------------------------------
// Photonic delay reservoir on gfx1250 (MI455X).
//   B=64, S=512, IN_DIM=4, R=1024, taps {1,4,24,96,168}, leak 0.1
// Strategy:
//   * per-step feedback GEMM (64x1024 @ 5x 1024x1024) via v_wmma_f32_16x16x32_f16
//   * 64 persistent workgroups, one 16-column N-tile each; f16 weight slice
//     (5 taps x 16 cols x 1024 K = 160KB) resident in LDS for all 512 steps
//   * global f16 ring buffer [256][64][1024] of past states (slots never
//     written stay zero == reference's zero history)
//   * h kept in registers (C/D fragment layout), tanh/leak fused, f32 out
//   * device-scope atomic arrive + spin barrier between timesteps
// ---------------------------------------------------------------------------

#define BATCH   64
#define SEQ     512
#define RDIM    1024
#define NTAPS   5
#define NWG     64          // R / 16 columns
#define RING_P  256         // ring slots (power of two > max delay 168)
#define SLOT_E  (BATCH * RDIM)          // elements per ring slot
#define LEAK    0.1f

typedef __attribute__((ext_vector_type(16))) _Float16 v16h;
typedef __attribute__((ext_vector_type(8)))  float    v8f;

union ABFrag { v16h v; uint4 q[2]; };

__device__ __constant__ int kTaps[NTAPS] = {1, 4, 24, 96, 168};

// ---------------------------------------------------------------------------
// Prep: Wt[((n*5)+tap)*1024 + k] = (f16) W_fb[tap][k][n]   (transpose+convert)
// ---------------------------------------------------------------------------
__global__ void prep_weights_kernel(const float* __restrict__ Wfb,
                                    _Float16* __restrict__ Wt) {
    unsigned i = blockIdx.x * 256u + threadIdx.x;      // over 5*1024*1024
    unsigned n   = i / (NTAPS * RDIM);
    unsigned rem = i % (NTAPS * RDIM);
    unsigned tap = rem / RDIM;
    unsigned k   = rem % RDIM;
    Wt[i] = (_Float16)Wfb[((size_t)tap * RDIM + k) * RDIM + n];
}

// ---------------------------------------------------------------------------
// Main persistent kernel: grid = 64 WGs x 256 threads (8 waves).
//   wave = khalf*4 + mtile : mtile in 0..3 (16 batches each),
//                            khalf in 0..1 (80 of 160 k-tiles each)
// ---------------------------------------------------------------------------
__launch_bounds__(256, 1)
__global__ void reservoir_kernel(const float* __restrict__ x,      // [64][512][4]
                                 const float* __restrict__ W_in,   // [1024][4]
                                 const float* __restrict__ bias,   // [1024]
                                 const _Float16* __restrict__ Wt,  // [1024][5][1024]
                                 _Float16* __restrict__ ring,      // [256][64][1024]
                                 float* __restrict__ out,          // [64][512][1024]
                                 unsigned* __restrict__ counter) {
    __shared__ _Float16 ldsW[16 * NTAPS * RDIM];   // 160 KB weight slice
    __shared__ float    ldsAcc[4][32][8];          // 4 KB partial reduction

    const int g    = blockIdx.x;           // N-tile id, columns [16g, 16g+16)
    const int tid  = threadIdx.x;
    const int wave = tid >> 5;
    const int lane = tid & 31;
    const int lgrp = lane >> 4;            // 0/1 : K sub-block select
    const int lrow = lane & 15;            // A: M row / B: N column
    const int n0   = g * 16;

    const int mtile = wave & 3;
    const int khalf = wave >> 2;

    // ---- load this WG's weight slice into LDS (rows n0..n0+15, all taps) ----
    {
        const uint4* src = (const uint4*)(Wt + (size_t)n0 * NTAPS * RDIM);
        uint4* dst = (uint4*)ldsW;
        const int nvec = 16 * NTAPS * RDIM / 8;        // 10240 x 16B
        for (int i = tid; i < nvec; i += 256) dst[i] = src[i];
    }
    __syncthreads();

    // ---- loop-invariant per-lane finalize constants (waves 0-3 use them) ----
    const int col = n0 + lrow;                          // output column
    const float w0 = W_in[col * 4 + 0];
    const float w1 = W_in[col * 4 + 1];
    const float w2 = W_in[col * 4 + 2];
    const float w3 = W_in[col * 4 + 3];
    const float bv = bias[col];
    float h[8];
    #pragma unroll
    for (int v = 0; v < 8; ++v) h[v] = 0.0f;

    // A fragment: row = mtile*16 + lrow, element offset 8*lgrp within k-tile
    const int aRowOff = (mtile * 16 + lrow) * RDIM + 8 * lgrp;

    const int ktLo = khalf * 80;       // this wave's k-tile range [ktLo, ktHi)
    const int ktHi = ktLo + 80;

    for (int t = 0; t < SEQ; ++t) {
        v8f acc = {};

        #pragma unroll
        for (int tap = 0; tap < NTAPS; ++tap) {
            const int tbase = tap * 32;                      // k-tiles of this tap
            int lo = ktLo - tbase; if (lo < 0)  lo = 0;
            int hi = ktHi - tbase; if (hi > 32) hi = 32;
            if (lo >= hi) continue;

            const int slot = (t - kTaps[tap]) & (RING_P - 1);
            const _Float16* abase = ring + (size_t)slot * SLOT_E + aRowOff;
            const _Float16* bbase = ldsW + (lrow * NTAPS + tap) * RDIM + 8 * lgrp;

            for (int kk = lo; kk < hi; ++kk) {
                ABFrag a, b;
                const _Float16* ap = abase + kk * 32;
                const _Float16* bp = bbase + kk * 32;
                a.q[0] = *(const uint4*)(ap);
                a.q[1] = *(const uint4*)(ap + 16);
                b.q[0] = *(const uint4*)(bp);
                b.q[1] = *(const uint4*)(bp + 16);
                acc = __builtin_amdgcn_wmma_f32_16x16x32_f16(
                          false, a.v, false, b.v, (short)0, acc, false, false);
            }
        }

        // ---- reduce the two K-halves through LDS ----
        if (khalf == 1) {
            #pragma unroll
            for (int v = 0; v < 8; ++v) ldsAcc[mtile][lane][v] = acc[v];
        }
        __syncthreads();

        if (khalf == 0) {
            _Float16* rw = ring + (size_t)(t & (RING_P - 1)) * SLOT_E;
            #pragma unroll
            for (int v = 0; v < 8; ++v) {
                const float fb = acc[v] + ldsAcc[mtile][lane][v];
                const int batch = mtile * 16 + v + 8 * lgrp;   // C/D layout row
                const float4 xv = *(const float4*)(x + ((size_t)batch * SEQ + t) * 4);
                const float drive = xv.x * w0 + xv.y * w1 + xv.z * w2 + xv.w * w3;
                const float val = tanhf(drive + fb + bv);
                const float hn = (1.0f - LEAK) * h[v] + LEAK * val;
                h[v] = hn;
                out[((size_t)batch * SEQ + t) * RDIM + col] = hn;
                rw[batch * RDIM + col] = (_Float16)hn;
            }
        }

        // ---- device-wide step barrier: release, arrive, spin, acquire ----
        __threadfence();
        __syncthreads();
        if (tid == 0) {
            __hip_atomic_fetch_add(counter, 1u, __ATOMIC_RELEASE,
                                   __HIP_MEMORY_SCOPE_AGENT);
            const unsigned target = (unsigned)(t + 1) * (unsigned)NWG;
            while (__hip_atomic_load(counter, __ATOMIC_ACQUIRE,
                                     __HIP_MEMORY_SCOPE_AGENT) < target) {
                __builtin_amdgcn_s_sleep(2);
            }
        }
        __syncthreads();
        __threadfence();
    }
}

// ---------------------------------------------------------------------------
// Launch: inputs per setup_inputs() order: x, W_in, W_fb, bias (all f32).
// Workspace layout: [0,1024) counter, [1KB, +10MB) Wt f16, then 32MB ring f16.
// ---------------------------------------------------------------------------
extern "C" void kernel_launch(void* const* d_in, const int* in_sizes, int n_in,
                              void* d_out, int out_size, void* d_ws, size_t ws_size,
                              hipStream_t stream) {
    const float* x    = (const float*)d_in[0];
    const float* W_in = (const float*)d_in[1];
    const float* W_fb = (const float*)d_in[2];
    const float* bias = (const float*)d_in[3];
    float* out        = (float*)d_out;

    unsigned char* ws = (unsigned char*)d_ws;
    unsigned* counter = (unsigned*)ws;
    _Float16* Wt      = (_Float16*)(ws + 1024);
    const size_t wtBytes   = (size_t)NTAPS * RDIM * RDIM * sizeof(_Float16); // 10 MB
    _Float16* ring    = (_Float16*)(ws + 1024 + wtBytes);
    const size_t ringBytes = (size_t)RING_P * SLOT_E * sizeof(_Float16);     // 32 MB

    hipMemsetAsync(counter, 0, 1024, stream);
    hipMemsetAsync(ring, 0, ringBytes, stream);

    const int wtElems = NTAPS * RDIM * RDIM;
    prep_weights_kernel<<<wtElems / 256, 256, 0, stream>>>(W_fb, Wt);

    reservoir_kernel<<<NWG, 256, 0, stream>>>(x, W_in, bias, Wt, ring, out, counter);
}
